// MeshGNN_85487029060318
// MI455X (gfx1250) — compile-verified, matrix operand order checked by reading, not run on hardware
//
#include <hip/hip_runtime.h>
#include <cstdint>
#include <cstddef>

#define EPSBN 1e-5f
#define H 256

typedef __bf16 bf16_t;
typedef __attribute__((ext_vector_type(16))) __bf16 v16bf;
typedef __attribute__((ext_vector_type(8)))  __bf16 v8bf;
typedef __attribute__((ext_vector_type(8)))  float  v8f;

#if defined(__gfx1250__) && __has_builtin(__builtin_amdgcn_global_load_async_to_lds_b128) && __has_builtin(__builtin_amdgcn_s_wait_asynccnt)
#define USE_ASYNC_LDS 1
#else
#define USE_ASYNC_LDS 0
#endif

__device__ __forceinline__ unsigned short f2bf(float f) {
    unsigned int u = __float_as_uint(f);
    u += 0x7FFFu + ((u >> 16) & 1u);      // round to nearest even
    return (unsigned short)(u >> 16);
}
__device__ __forceinline__ float bf2f(unsigned short b) {
    return __uint_as_float(((unsigned int)b) << 16);
}

// ---------------- degree / pooling counts ----------------
__global__ void k_deg(const int* __restrict__ col, float* __restrict__ deg, int E) {
    int i = blockIdx.x * blockDim.x + threadIdx.x;
    if (i < E) atomicAdd(&deg[col[i]], 1.0f);
}
__global__ void k_cnt(const int* __restrict__ batch, float* __restrict__ cnt, int N) {
    int i = blockIdx.x * blockDim.x + threadIdx.x;
    if (i < N) atomicAdd(&cnt[batch[i]], 1.0f);
}
__global__ void k_dinv(float* __restrict__ d, int N) {
    int i = blockIdx.x * blockDim.x + threadIdx.x;
    if (i < N) d[i] = rsqrtf(d[i] + 1.0f);
}

// ---------------- BN0 over concat(pos, norm) ----------------
__global__ void k_bn0_stats(const float* __restrict__ pos, const float* __restrict__ nrm,
                            float* __restrict__ st, int N) {
    int gid = blockIdx.x * blockDim.x + threadIdx.x;
    int stride = gridDim.x * blockDim.x;
    float s[6] = {0,0,0,0,0,0}, q[6] = {0,0,0,0,0,0};
    for (int n = gid; n < N; n += stride) {
        #pragma unroll
        for (int k = 0; k < 3; ++k) {
            float a = pos[n*3+k]; s[k]   += a; q[k]   += a*a;
            float b = nrm[n*3+k]; s[3+k] += b; q[3+k] += b*b;
        }
    }
    #pragma unroll
    for (int k = 0; k < 6; ++k) {
        for (int off = 16; off > 0; off >>= 1) {
            s[k] += __shfl_xor(s[k], off, 32);
            q[k] += __shfl_xor(q[k], off, 32);
        }
    }
    if ((threadIdx.x & 31) == 0) {
        #pragma unroll
        for (int k = 0; k < 6; ++k) { atomicAdd(&st[k], s[k]); atomicAdd(&st[6+k], q[k]); }
    }
}
__global__ void k_bn0_apply(const float* __restrict__ pos, const float* __restrict__ nrm,
                            const float* __restrict__ st, const float* __restrict__ g,
                            const float* __restrict__ b, float* __restrict__ x0,
                            float invN, int N) {
    int idx = blockIdx.x * blockDim.x + threadIdx.x;
    if (idx >= N * 6) return;
    int n = idx / 6, k = idx - n * 6;
    float m  = st[k] * invN;
    float vr = st[6+k] * invN - m * m;
    float is = rsqrtf(vr + EPSBN);
    float v  = (k < 3) ? pos[n*3+k] : nrm[n*3+(k-3)];
    x0[idx] = (v - m) * is * g[k] + b[k];
}

// ---------------- layer 1 linear (6 -> 256), VALU is fine here ----------------
__global__ void k_lin6(const float* __restrict__ x0, const float* __restrict__ W,
                       unsigned short* __restrict__ xw, int N) {
    int idx = blockIdx.x * blockDim.x + threadIdx.x;
    if (idx >= N * H) return;
    int n = idx >> 8, ch = idx & 255;
    float s = 0.f;
    #pragma unroll
    for (int k = 0; k < 6; ++k) s += x0[n*6+k] * W[k*H+ch];
    xw[idx] = f2bf(s);
}

// ---------------- pack W[256x256] f32 -> WMMA B-fragment blocks (bf16) ----------------
// block bi = kb*16+nt; within block: lane l (0..31), elem e (0..15) contiguous,
// n = nt*16 + (l&15); h = l>>4; K = kb*32 + (e<8 ? 8h+e : 16+8h+(e-8))
__global__ void k_pack(const float* __restrict__ W, unsigned short* __restrict__ wpk) {
    int idx = blockIdx.x * blockDim.x + threadIdx.x;   // 0 .. 65535
    int bi = idx >> 9, le = idx & 511;
    int l = le >> 4, e = le & 15;
    int kb = bi >> 4, nt = bi & 15;
    int n  = nt * 16 + (l & 15);
    int hh = l >> 4;
    int k  = kb * 32 + ((e < 8) ? (8*hh + e) : (16 + 8*hh + (e - 8)));
    wpk[idx] = f2bf(W[k * H + n]);
}

// ---------------- WMMA GEMM: [N,256](bf16) x [256,256](packed bf16) -> [N,256](bf16) --------
// 256 threads = 8 waves; wave computes a 16x64 output stripe (4 accum tiles).
// B slab for each K-step (16 KB) is staged block-wide into LDS with async copies
// (double-buffered, overlapped with WMMA) when the toolchain exposes the builtins.
__global__ __launch_bounds__(256)
void k_gemm256(const unsigned short* __restrict__ xin,
               const unsigned short* __restrict__ wpk,
               unsigned short* __restrict__ xw, int nrows, int ntiles) {
    const int lane = threadIdx.x & 31;
    const int wave = threadIdx.x >> 5;                  // 0..7
    const int mtile = blockIdx.x * 2 + (wave >> 2);
    const int nstripe = wave & 3;                       // 4 stripes of 64 cols
    const int hh = lane >> 4;
    int m = mtile * 16 + (lane & 15);
    if (m >= nrows) m = nrows - 1;                      // safe read clamp
    const unsigned short* arow = xin + (size_t)m * H;

    v8f acc[4] = {v8f{}, v8f{}, v8f{}, v8f{}};
    union AV { v16bf v; v8bf h[2]; };

#if USE_ASYNC_LDS
    typedef int vec4i __attribute__((vector_size(16)));
    typedef __attribute__((address_space(3))) vec4i* lds4_t;
    __shared__ unsigned short sB[2][8192];              // 2 x 16KB K-slabs
    const int tid = threadIdx.x;
    auto stage = [&](int kb, int buf) {
        const unsigned short* src = wpk + (size_t)kb * 8192 + tid * 32;
        unsigned short* dst = &sB[buf][tid * 32];
        #pragma unroll
        for (int j = 0; j < 4; ++j) {
            __builtin_amdgcn_global_load_async_to_lds_b128(
                (vec4i*)(src + j * 8),
                (lds4_t)(dst + j * 8), 0, 0);
        }
    };
    stage(0, 0);
    __builtin_amdgcn_s_wait_asynccnt(0);
    __syncthreads();
#endif

    #pragma unroll
    for (int kb = 0; kb < 8; ++kb) {
#if USE_ASYNC_LDS
        if (kb + 1 < 8) stage(kb + 1, (kb + 1) & 1);    // overlap next slab with WMMAs
#endif
        AV a;
        a.h[0] = *(const v8bf*)(arow + kb*32 + 8*hh);        // K elems 0..7
        a.h[1] = *(const v8bf*)(arow + kb*32 + 16 + 8*hh);   // K elems 8..15
        #pragma unroll
        for (int t = 0; t < 4; ++t) {
            const int nt = nstripe * 4 + t;
#if USE_ASYNC_LDS
            const v16bf b = *(const v16bf*)(&sB[kb & 1][nt * 512 + lane * 16]);
#else
            const v16bf b = *(const v16bf*)(wpk + ((size_t)(kb*16 + nt) << 9) + lane*16);
#endif
            acc[t] = __builtin_amdgcn_wmma_f32_16x16x32_bf16(
                        false, a.v, false, b, (short)0, acc[t], false, false);
        }
#if USE_ASYNC_LDS
        __builtin_amdgcn_s_wait_asynccnt(0);
        __syncthreads();
#endif
    }
    // C layout: VGPR r, lanes 0-15: M=r ; lanes 16-31: M=8+r ; N = lane&15
    const int cb = nstripe * 64 + (lane & 15);
    const int rb = mtile * 16 + 8 * hh;
    #pragma unroll
    for (int t = 0; t < 4; ++t) {
        #pragma unroll
        for (int r = 0; r < 8; ++r) {
            int rr = rb + r;
            if (rr < nrows) xw[(size_t)rr * H + cb + t*16] = f2bf(acc[t][r]);
        }
    }
}

// ---------------- edge aggregation: one wave per edge, lane handles 8 channels ----------
__global__ void k_edge_agg(const int* __restrict__ row, const int* __restrict__ col,
                           const float* __restrict__ dinv,
                           const unsigned short* __restrict__ xw,
                           float* __restrict__ agg, int E) {
    long long gt = (long long)blockIdx.x * blockDim.x + threadIdx.x;
    int e = (int)(gt >> 5);
    int lane = threadIdx.x & 31;
    if (e >= E) return;
    int r = row[e], c = col[e];
    float w = dinv[r] * dinv[c];
    if (e + 64 < E) {   // pull a future source row L2 -> L0 (global_prefetch_b8)
        int rn = row[e + 64];
        __builtin_prefetch(xw + (size_t)rn * H + lane * 8, 0, 1);
    }
    union { uint4 v; unsigned short s[8]; } p;
    p.v = *(const uint4*)(xw + (size_t)r * H + lane * 8);
    float* dst = agg + (size_t)c * H + lane * 8;
    #pragma unroll
    for (int i = 0; i < 8; ++i) atomicAdd(dst + i, w * bf2f(p.s[i]));
}

// ---------------- h = agg + sl*xw + bias, relu, BN stats (in-place) ----------------
__global__ void k_postagg(float* __restrict__ h, const unsigned short* __restrict__ xw,
                          const float* __restrict__ dinv, const float* __restrict__ bias,
                          float* __restrict__ ssum, float* __restrict__ sss, int N) {
    int ch = threadIdx.x;
    int base = blockIdx.x * 64;
    float b = bias[ch];
    float s = 0.f, q = 0.f;
    for (int i = 0; i < 64; ++i) {
        int n = base + i;
        if (n >= N) break;
        size_t idx = (size_t)n * H + ch;
        float d = dinv[n];
        float v = h[idx] + d * d * bf2f(xw[idx]) + b;
        v = fmaxf(v, 0.f);
        h[idx] = v;
        s += v; q += v * v;
    }
    atomicAdd(&ssum[ch], s);
    atomicAdd(&sss[ch], q);
}

__global__ void k_bnfin(const float* __restrict__ ssum, const float* __restrict__ sss,
                        const float* __restrict__ g, const float* __restrict__ be,
                        float* __restrict__ A, float* __restrict__ B, float invN) {
    int ch = threadIdx.x;
    float m  = ssum[ch] * invN;
    float vr = sss[ch] * invN - m * m;
    float a  = g[ch] * rsqrtf(vr + EPSBN);
    A[ch] = a;
    B[ch] = be[ch] - m * a;
}

__global__ void k_bnapply(const float* __restrict__ h, const float* __restrict__ A,
                          const float* __restrict__ B, unsigned short* __restrict__ xout,
                          int N) {
    int idx = blockIdx.x * blockDim.x + threadIdx.x;
    if (idx >= N * H) return;
    int ch = idx & 255;
    xout[idx] = f2bf(h[idx] * A[ch] + B[ch]);
}

__global__ void k_bnapply_pool(const float* __restrict__ h, const float* __restrict__ A,
                               const float* __restrict__ B, const int* __restrict__ batch,
                               float* __restrict__ out, int N) {
    int idx = blockIdx.x * blockDim.x + threadIdx.x;
    if (idx >= N * H) return;
    int n = idx >> 8, ch = idx & 255;
    float y = h[idx] * A[ch] + B[ch];
    atomicAdd(&out[(size_t)batch[n] * H + ch], y);
}

__global__ void k_finalize(float* __restrict__ out, const float* __restrict__ cnt, int G) {
    int idx = blockIdx.x * blockDim.x + threadIdx.x;
    if (idx >= G * H) return;
    out[idx] /= fmaxf(cnt[idx >> 8], 1.0f);
}

// =================================================================================
extern "C" void kernel_launch(void* const* d_in, const int* in_sizes, int n_in,
                              void* d_out, int out_size, void* d_ws, size_t ws_size,
                              hipStream_t stream) {
    const float* pos   = (const float*)d_in[0];
    const float* nrm   = (const float*)d_in[1];
    const int*   eidx  = (const int*)  d_in[2];     // [2,E]
    const int*   batch = (const int*)  d_in[3];
    const float* bn0g  = (const float*)d_in[4];
    const float* bn0b  = (const float*)d_in[5];

    const int N = in_sizes[0] / 3;
    const int E = in_sizes[2] / 2;
    const int G = out_size / H;
    const int* erow = eidx;
    const int* ecol = eidx + E;
    float* out = (float*)d_out;

    // ---- workspace carve-up (256B aligned) ----
    char* ws = (char*)d_ws;
    size_t off = 0;
    auto carve = [&](size_t bytes) { size_t o = off; off = (off + bytes + 255) & ~(size_t)255; return o; };
    float*          dinv = (float*)         (ws + carve((size_t)N * 4));
    float*          x0   = (float*)         (ws + carve((size_t)N * 6 * 4));
    unsigned short* xw   = (unsigned short*)(ws + carve((size_t)N * H * 2));
    float*          agg  = (float*)         (ws + carve((size_t)N * H * 4));
    unsigned short* xA   = (unsigned short*)(ws + carve((size_t)N * H * 2));
    unsigned short* xB   = (unsigned short*)(ws + carve((size_t)N * H * 2));
    unsigned short* wpk  = (unsigned short*)(ws + carve((size_t)H * H * 2));
    float*          ssum = (float*)         (ws + carve(2 * H * 4));   // ssum | sss contiguous
    float*          sss  = ssum + H;
    float*          Abuf = (float*)         (ws + carve(H * 4));
    float*          Bbuf = (float*)         (ws + carve(H * 4));
    float*          bn0s = (float*)         (ws + carve(16 * 4));
    float*          cnt  = (float*)         (ws + carve((size_t)G * 4));
    (void)ws_size; (void)n_in;

    const float invN = 1.0f / (float)N;
    const int ntiles = (N + 15) / 16;

    // ---- degree norm + pool counts ----
    (void)hipMemsetAsync(dinv, 0, (size_t)N * 4, stream);
    (void)hipMemsetAsync(cnt,  0, (size_t)G * 4, stream);
    k_deg<<<(E + 255) / 256, 256, 0, stream>>>(ecol, dinv, E);
    k_cnt<<<(N + 255) / 256, 256, 0, stream>>>(batch, cnt, N);
    k_dinv<<<(N + 255) / 256, 256, 0, stream>>>(dinv, N);

    // ---- BN0 ----
    (void)hipMemsetAsync(bn0s, 0, 16 * 4, stream);
    k_bn0_stats<<<64, 256, 0, stream>>>(pos, nrm, bn0s, N);
    k_bn0_apply<<<(N * 6 + 255) / 256, 256, 0, stream>>>(pos, nrm, bn0s, bn0g, bn0b, x0, invN, N);

    // ---- 4 GCN layers ----
    unsigned short* xin  = xA;
    unsigned short* xout = xA;
    for (int l = 1; l <= 4; ++l) {
        const int ii = 6 + (l - 1) * 4;
        const float* W    = (const float*)d_in[ii + 0];
        const float* bias = (const float*)d_in[ii + 1];
        const float* gg   = (const float*)d_in[ii + 2];
        const float* be   = (const float*)d_in[ii + 3];

        if (l == 1) {
            k_lin6<<<(N * H + 255) / 256, 256, 0, stream>>>(x0, W, xw, N);
        } else {
            k_pack<<<H * H / 256, 256, 0, stream>>>(W, wpk);
            k_gemm256<<<(ntiles + 1) / 2, 256, 0, stream>>>(xin, wpk, xw, N, ntiles);
        }

        (void)hipMemsetAsync(agg,  0, (size_t)N * H * 4, stream);
        (void)hipMemsetAsync(ssum, 0, 2 * H * 4, stream);
        k_edge_agg<<<(E + 7) / 8, 256, 0, stream>>>(erow, ecol, dinv, xw, agg, E);
        k_postagg<<<(N + 63) / 64, 256, 0, stream>>>(agg, xw, dinv, bias, ssum, sss, N);
        k_bnfin<<<1, 256, 0, stream>>>(ssum, sss, gg, be, Abuf, Bbuf, invN);

        if (l < 4) {
            xout = (l & 1) ? xA : xB;
            k_bnapply<<<(N * H + 255) / 256, 256, 0, stream>>>(agg, Abuf, Bbuf, xout, N);
            xin = xout;
        } else {
            (void)hipMemsetAsync(out, 0, (size_t)out_size * 4, stream);
            k_bnapply_pool<<<(N * H + 255) / 256, 256, 0, stream>>>(agg, Abuf, Bbuf, batch, out, N);
            k_finalize<<<(G * H + 255) / 256, 256, 0, stream>>>(out, cnt, G);
        }
    }
}
